// FEDformerEncoder_36086315221240
// MI455X (gfx1250) — matrix operand order repaired and checked by reading, not compiled, not run on hardware
//
#include <hip/hip_runtime.h>
#include <hip/hip_bf16.h>

// Problem constants
#define BB    16
#define LL    3000
#define CIN   12
#define DDIM  512
#define HH    8
#define EE    64
#define MMODE 64
#define DFF   2048
#define NLAYER 2
#define NCLS  5
#define LPAD  3008      // 3000 padded to multiple of 32 for WMMA K-loop
#define NM2   128       // 64 real + 64 imag modes stacked

typedef __attribute__((ext_vector_type(16))) __bf16 v16bf;
typedef __attribute__((ext_vector_type(8)))  float  v8f;

union V16BF { uint4 q[2]; unsigned u[8]; v16bf v; };

// ---------------------------------------------------------------------------
// WMMA GEMM, bf16 operands in memory: C[Mr, nstore] = act(A[Mr,K] @ W[N,K]^T + b)
// One wave computes a 16(M) x 64(N) strip: 4 accumulators, 4 WMMAs per K-step.
// A-fragment loads: 2 x b128 ; B-fragments: 8 x b128 ; zero conversion VALU.
// bf16 pairs in memory already match the WMMA dword packing (little-endian).
// ---------------------------------------------------------------------------
__global__ __launch_bounds__(256)
void gemm_bf16_wmma(const __bf16* __restrict__ A,
                    const __bf16* __restrict__ W,
                    const float* __restrict__ bias,
                    void* __restrict__ Cv,
                    int Mr, int N, int K, int ldc,
                    int nstore, int relu, int out_bf16)
{
  const int gtid   = blockIdx.x * blockDim.x + threadIdx.x;
  const int wave   = gtid >> 5;
  const int lane   = threadIdx.x & 31;
  const int tilesN = N >> 6;                       // 64-wide column groups
  const int tilesM = Mr >> 4;
  if (wave >= tilesM * tilesN) return;             // wave-uniform: EXEC all-1 inside
  const int tm = wave / tilesN;
  const int tn = wave % tilesN;
  const int l15  = lane & 15;
  const int half = lane >> 4;

  const __bf16* Arow = A + (size_t)(tm * 16 + l15) * K;
  const __bf16* Wr0  = W + (size_t)(tn * 64 +  0 + l15) * K;
  const __bf16* Wr1  = W + (size_t)(tn * 64 + 16 + l15) * K;
  const __bf16* Wr2  = W + (size_t)(tn * 64 + 32 + l15) * K;
  const __bf16* Wr3  = W + (size_t)(tn * 64 + 48 + l15) * K;

  v8f acc0 = {}, acc1 = {}, acc2 = {}, acc3 = {};

  for (int k0 = 0; k0 < K; k0 += 32) {
    // A layout: lanes hold rows 0..15; V0..7 hold K = half*16 + 0..15 (contiguous, matches B)
    V16BF av;
    av.q[0] = *(const uint4*)(Arow + k0 + half * 16);
    av.q[1] = *(const uint4*)(Arow + k0 + half * 16 + 8);

    // B layout: lane n = l15; V0..7 hold K = half*16 + 0..15 (contiguous)
    const int kb = k0 + half * 16;
    V16BF b0, b1, b2, b3;
    b0.q[0] = *(const uint4*)(Wr0 + kb); b0.q[1] = *(const uint4*)(Wr0 + kb + 8);
    b1.q[0] = *(const uint4*)(Wr1 + kb); b1.q[1] = *(const uint4*)(Wr1 + kb + 8);
    b2.q[0] = *(const uint4*)(Wr2 + kb); b2.q[1] = *(const uint4*)(Wr2 + kb + 8);
    b3.q[0] = *(const uint4*)(Wr3 + kb); b3.q[1] = *(const uint4*)(Wr3 + kb + 8);

    __builtin_prefetch(Arow + k0 + 256, 0, 1);   // global_prefetch next chunks
    __builtin_prefetch(Wr0 + kb + 256, 0, 1);

    acc0 = __builtin_amdgcn_wmma_f32_16x16x32_bf16(false, av.v, false, b0.v,
                                                   (short)0, acc0, false, false);
    acc1 = __builtin_amdgcn_wmma_f32_16x16x32_bf16(false, av.v, false, b1.v,
                                                   (short)0, acc1, false, false);
    acc2 = __builtin_amdgcn_wmma_f32_16x16x32_bf16(false, av.v, false, b2.v,
                                                   (short)0, acc2, false, false);
    acc3 = __builtin_amdgcn_wmma_f32_16x16x32_bf16(false, av.v, false, b3.v,
                                                   (short)0, acc3, false, false);
  }

  // C/D layout: lane half selects M offset 8; VGPR r -> row r; N = lane&15
  const int row0 = tm * 16 + half * 8;
  v8f accs[4] = {acc0, acc1, acc2, acc3};
#pragma unroll
  for (int j = 0; j < 4; ++j) {
    const int colg = tn * 64 + j * 16 + l15;
    if (colg < nstore) {
      const float bv = bias ? bias[colg] : 0.f;
#pragma unroll
      for (int r = 0; r < 8; ++r) {
        float v = accs[j][r] + bv;
        if (relu) v = fmaxf(v, 0.f);
        const size_t ci = (size_t)(row0 + r) * ldc + colg;
        if (out_bf16) ((__bf16*)Cv)[ci] = (__bf16)v;
        else          ((float*)Cv)[ci]  = v;
      }
    }
  }
}

// ---------------------------------------------------------------------------
// fp32 -> bf16 bulk convert
// ---------------------------------------------------------------------------
__global__ void cvt_bf16_kernel(__bf16* __restrict__ dst,
                                const float* __restrict__ src, long n)
{
  long i = (long)blockIdx.x * blockDim.x + threadIdx.x;
  if (i < n) dst[i] = (__bf16)src[i];
}

// ---------------------------------------------------------------------------
// DataEmbedding_onlypos: circular conv1d(k=3, no bias) + sinusoidal PE
// ---------------------------------------------------------------------------
__global__ void embed_kernel(const float* __restrict__ xe,
                             const float* __restrict__ ew,
                             float* __restrict__ X)
{
  int idx = blockIdx.x * blockDim.x + threadIdx.x;
  if (idx >= BB * LL * DDIM) return;
  int d  = idx & (DDIM - 1);
  int bt = idx >> 9;
  int t  = bt % LL;
  int b  = bt / LL;
  float acc = 0.f;
#pragma unroll
  for (int j = 0; j < 3; ++j) {
    int tt = t + j - 1;
    tt = (tt < 0) ? tt + LL : (tt >= LL ? tt - LL : tt);
    const float* xr = xe + ((size_t)b * LL + tt) * CIN;
    const float* wf = ew + (size_t)d * (CIN * 3) + j;
#pragma unroll
    for (int c = 0; c < CIN; ++c) acc += xr[c] * wf[c * 3];
  }
  int   i   = d >> 1;
  float dv  = __expf(-logf(10000.f) * (2.f * (float)i) / (float)DDIM);
  float ang = (float)t * dv;
  float pe  = (d & 1) ? __cosf(ang) : __sinf(ang);
  X[idx] = acc + pe;
}

// ---------------------------------------------------------------------------
// Transpose q[B,L,D](bf16) -> qT[(b*D+d), t](bf16) padded to LPAD with zeros
// ---------------------------------------------------------------------------
__global__ void qtrans_kernel(const __bf16* __restrict__ q, __bf16* __restrict__ qT)
{
  long idx = (long)blockIdx.x * blockDim.x + threadIdx.x;
  if (idx >= (long)BB * DDIM * LPAD) return;
  int t = (int)(idx % LPAD);
  int r = (int)(idx / LPAD);
  int d = r & (DDIM - 1);
  int b = r >> 9;
  qT[idx] = (t < LL) ? q[((size_t)b * LL + t) * DDIM + d] : (__bf16)0.f;
}

// ---------------------------------------------------------------------------
// Forward DFT basis WF[128, LPAD] (bf16): rows 0..63 = cos, 64..127 = -sin
// ---------------------------------------------------------------------------
__global__ void gen_fwd_basis(__bf16* __restrict__ WF)
{
  int idx = blockIdx.x * blockDim.x + threadIdx.x;
  if (idx >= NM2 * LPAD) return;
  int t  = idx % LPAD;
  int m2 = idx / LPAD;
  float v = 0.f;
  if (t < LL) {
    int   m   = m2 & 63;
    int   red = (m * t) % LL;
    float ang = 6.2831853071795864769f * (float)red / (float)LL;
    v = (m2 < 64) ? __cosf(ang) : -__sinf(ang);
  }
  WF[idx] = (__bf16)v;
}

// ---------------------------------------------------------------------------
// Inverse DFT basis WI[LPAD, 128] (bf16): irfft with only modes 0..63 nonzero
//   x[t] = Yr[0]/L + sum_{m>=1} (2/L)(Yr[m] cos - Yi[m] sin)
// ---------------------------------------------------------------------------
__global__ void gen_inv_basis(__bf16* __restrict__ WI)
{
  int idx = blockIdx.x * blockDim.x + threadIdx.x;
  if (idx >= LPAD * NM2) return;
  int m2 = idx % NM2;
  int t  = idx / NM2;
  float v = 0.f;
  if (t < LL) {
    int   m   = m2 & 63;
    float c   = (m == 0) ? (1.f / (float)LL) : (2.f / (float)LL);
    int   red = (m * t) % LL;
    float ang = 6.2831853071795864769f * (float)red / (float)LL;
    if (m2 < 64)      v = c * __cosf(ang);
    else if (m != 0)  v = -c * __sinf(ang);   // imag of mode 0 ignored by irfft
  }
  WI[idx] = (__bf16)v;
}

// ---------------------------------------------------------------------------
// Complex mode mixing: Y[b,h,o,m] = sum_e X[b,h,e,m] * w[h,e,o,m]
// Xft fp32; Y2 written bf16 (feeds inverse-DFT WMMA GEMM)
// ---------------------------------------------------------------------------
__global__ void modemix_kernel(const float* __restrict__ Xft,
                               const float* __restrict__ wr,
                               const float* __restrict__ wi,
                               __bf16* __restrict__ Y2)
{
  int idx = blockIdx.x * blockDim.x + threadIdx.x;
  if (idx >= BB * HH * EE * MMODE) return;
  int m = idx & 63;
  int o = (idx >> 6) & 63;
  int h = (idx >> 12) & 7;
  int b = idx >> 15;
  float yr = 0.f, yi = 0.f;
  const float* xbase = Xft + ((size_t)b * DDIM + h * EE) * NM2;
  const float* wrb   = wr + (((size_t)h * EE) * EE + o) * MMODE + m;
  const float* wib   = wi + (((size_t)h * EE) * EE + o) * MMODE + m;
  for (int e = 0; e < EE; ++e) {
    float xr = xbase[e * NM2 + m];
    float xi = xbase[e * NM2 + 64 + m];
    float wrv = wrb[(size_t)e * EE * MMODE];
    float wiv = wib[(size_t)e * EE * MMODE];
    yr += xr * wrv - xi * wiv;
    yi += xr * wiv + xi * wrv;
  }
  size_t row = (size_t)b * DDIM + h * EE + o;
  Y2[row * NM2 + m]      = (__bf16)yr;
  Y2[row * NM2 + 64 + m] = (__bf16)yi;
}

// ---------------------------------------------------------------------------
// Elementwise add + series-decomposition residual (x - mov_avg(x), k=25,
// replicate padding == index clamping)
// ---------------------------------------------------------------------------
__global__ void add_kernel(float* __restrict__ out, const float* __restrict__ a,
                           const float* __restrict__ b)
{
  int idx = blockIdx.x * blockDim.x + threadIdx.x;
  if (idx >= BB * LL * DDIM) return;
  out[idx] = a[idx] + b[idx];
}

__global__ void decomp_kernel(float* __restrict__ out, const float* __restrict__ s)
{
  int idx = blockIdx.x * blockDim.x + threadIdx.x;
  if (idx >= BB * LL * DDIM) return;
  int d  = idx & (DDIM - 1);
  int bt = idx >> 9;
  int t  = bt % LL;
  int b  = bt / LL;
  const float* base = s + (size_t)b * LL * DDIM + d;
  float sum = 0.f;
#pragma unroll
  for (int j = -12; j <= 12; ++j) {
    int tt = t + j;
    tt = tt < 0 ? 0 : (tt > LL - 1 ? LL - 1 : tt);
    sum += base[(size_t)tt * DDIM];
  }
  out[idx] = s[idx] - sum * (1.f / 25.f);
}

// ---------------------------------------------------------------------------
// Mean over L, then tiny classifier head
// ---------------------------------------------------------------------------
__global__ void pool_kernel(const float* __restrict__ X, float* __restrict__ P)
{
  int idx = blockIdx.x * blockDim.x + threadIdx.x;
  if (idx >= BB * DDIM) return;
  int d = idx & (DDIM - 1);
  int b = idx >> 9;
  const float* base = X + (size_t)b * LL * DDIM + d;
  float s = 0.f;
  for (int t = 0; t < LL; ++t) s += base[(size_t)t * DDIM];
  P[idx] = s * (1.f / (float)LL);
}

__global__ void head_kernel(const float* __restrict__ P,
                            const float* __restrict__ hw,
                            const float* __restrict__ hb,
                            float* __restrict__ out)
{
  int idx = blockIdx.x * blockDim.x + threadIdx.x;
  if (idx >= BB * NCLS) return;
  int c = idx % NCLS;
  int b = idx / NCLS;
  float s = hb[c];
  const float* pr = P + (size_t)b * DDIM;
  const float* wr = hw + (size_t)c * DDIM;
  for (int d = 0; d < DDIM; ++d) s += pr[d] * wr[d];
  out[idx] = s;
}

// ---------------------------------------------------------------------------
static inline void launch_gemm(const __bf16* A, const __bf16* W, const float* bias,
                               void* C, int Mr, int N, int K, int ldc, int nstore,
                               int relu, int out_bf16, hipStream_t s)
{
  long waves   = (long)(Mr >> 4) * (long)(N >> 6);   // 16 x 64 strip per wave
  long threads = waves * 32;
  int  blocks  = (int)((threads + 255) / 256);
  gemm_bf16_wmma<<<blocks, 256, 0, s>>>(A, W, bias, C, Mr, N, K, ldc,
                                        nstore, relu, out_bf16);
}

static inline void launch_cvt(__bf16* dst, const float* src, long n, hipStream_t s)
{
  cvt_bf16_kernel<<<(int)((n + 255) / 256), 256, 0, s>>>(dst, src, n);
}

extern "C" void kernel_launch(void* const* d_in, const int* in_sizes, int n_in,
                              void* d_out, int out_size, void* d_ws, size_t ws_size,
                              hipStream_t stream)
{
  (void)in_sizes; (void)n_in; (void)out_size; (void)ws_size;
  const float* x_enc = (const float*)d_in[0];
  const float* emb_w = (const float*)d_in[1];
  const float* Wq    = (const float*)d_in[2];
  const float* bq    = (const float*)d_in[3];
  // d_in[4..7] (Wk,bk,Wv,bv) are dead code in the reference output
  const float* Wo    = (const float*)d_in[8];
  const float* bo    = (const float*)d_in[9];
  const float* fwr   = (const float*)d_in[10];
  const float* fwi   = (const float*)d_in[11];
  const float* W1    = (const float*)d_in[12];
  const float* W2    = (const float*)d_in[13];
  const float* hw    = (const float*)d_in[14];
  const float* hb    = (const float*)d_in[15];
  float* out = (float*)d_out;

  // --- workspace carving (256B-aligned so all b128 fragment loads align) ---
  char* p = (char*)d_ws;
  auto carve = [&](size_t bytes) -> char* {
    char* r = p;
    p += (bytes + 255) & ~(size_t)255;
    return r;
  };
  const size_t NX = (size_t)BB * LL * DDIM;          // 24.576M elems

  float*  X     = (float*)carve(NX * 4);             // residual stream (fp32)
  float*  BUF1f = (float*)carve(NX * 4);
  float*  BUF2f = (float*)carve(NX * 4);
  __bf16* XBF   = (__bf16*)carve(NX * 2);            // bf16 copy of X for GEMMs
  __bf16* QBF   = (__bf16*)carve(NX * 2);            // q / attn (disjoint lifetimes)
  __bf16* QTBF  = (__bf16*)carve((size_t)BB * DDIM * LPAD * 2);
  __bf16* HIDBF = (__bf16*)carve((size_t)BB * LL * DFF * 2);
  float*  XFT   = (float*)carve((size_t)BB * DDIM * NM2 * 4);
  __bf16* Y2BF  = (__bf16*)carve((size_t)BB * DDIM * NM2 * 2);
  __bf16* WFBF  = (__bf16*)carve((size_t)NM2 * LPAD * 2);
  __bf16* WIBF  = (__bf16*)carve((size_t)LPAD * NM2 * 2);
  __bf16* WqBF  = (__bf16*)carve((size_t)NLAYER * DDIM * DDIM * 2);
  __bf16* WoBF  = (__bf16*)carve((size_t)NLAYER * DDIM * DDIM * 2);
  __bf16* W1BF  = (__bf16*)carve((size_t)NLAYER * DFF * DDIM * 2);
  __bf16* W2BF  = (__bf16*)carve((size_t)NLAYER * DDIM * DFF * 2);
  float*  POOL  = (float*)carve((size_t)BB * DDIM * 4);

  const int NE = BB * LL * DDIM;

  // DFT bases + weight conversion (regenerated every call: deterministic)
  gen_fwd_basis<<<(NM2 * LPAD + 255) / 256, 256, 0, stream>>>(WFBF);
  gen_inv_basis<<<(LPAD * NM2 + 255) / 256, 256, 0, stream>>>(WIBF);
  launch_cvt(WqBF, Wq, (long)NLAYER * DDIM * DDIM, stream);
  launch_cvt(WoBF, Wo, (long)NLAYER * DDIM * DDIM, stream);
  launch_cvt(W1BF, W1, (long)NLAYER * DFF * DDIM, stream);
  launch_cvt(W2BF, W2, (long)NLAYER * DDIM * DFF, stream);

  // Embedding + positional encoding -> X [B,L,D]
  embed_kernel<<<(NE + 255) / 256, 256, 0, stream>>>(x_enc, emb_w, X);

  for (int l = 0; l < NLAYER; ++l) {
    const __bf16* Wq_l = WqBF + (size_t)l * DDIM * DDIM;
    const float*  bq_l = bq   + (size_t)l * DDIM;
    const __bf16* Wo_l = WoBF + (size_t)l * DDIM * DDIM;
    const float*  bo_l = bo   + (size_t)l * DDIM;
    const float*  wr_l = fwr  + (size_t)l * HH * EE * EE * MMODE;
    const float*  wi_l = fwi  + (size_t)l * HH * EE * EE * MMODE;
    const __bf16* W1_l = W1BF + (size_t)l * DFF * DDIM;
    const __bf16* W2_l = W2BF + (size_t)l * DDIM * DFF;

    // q = X @ Wq^T + bq  -> bf16 directly (only qtrans consumes it)
    launch_cvt(XBF, X, NE, stream);
    launch_gemm(XBF, Wq_l, bq_l, QBF, BB * LL, DDIM, DDIM, DDIM, DDIM,
                0, 1, stream);

    // transpose to [ (b,d), t ] with zero-padded tail (K mult of 32)
    {
      long n = (long)BB * DDIM * LPAD;
      qtrans_kernel<<<(int)((n + 255) / 256), 256, 0, stream>>>(QBF, QTBF);
    }

    // forward DFT (modes 0..63, real|imag)   [8192, 128]   WMMA, K=3008
    launch_gemm(QTBF, WFBF, nullptr, XFT, BB * DDIM, NM2, LPAD, NM2, NM2,
                0, 0, stream);

    // per-head complex 64x64 mode mixing -> bf16
    modemix_kernel<<<(BB * HH * EE * MMODE + 255) / 256, 256, 0, stream>>>(
        XFT, wr_l, wi_l, Y2BF);

    // inverse DFT -> attn [B,H,E,L] contiguous (== view [B,L,D]) -> bf16
    launch_gemm(Y2BF, WIBF, nullptr, QBF, BB * DDIM, LPAD, NM2, LL, LL,
                0, 1, stream);

    // new_x = attn @ Wo^T + bo               [48000, 512]  -> fp32
    launch_gemm(QBF, Wo_l, bo_l, BUF2f, BB * LL, DDIM, DDIM, DDIM, DDIM,
                0, 0, stream);

    // x = (x + new_x) - mov_avg(x + new_x)
    add_kernel<<<(NE + 255) / 256, 256, 0, stream>>>(BUF1f, X, BUF2f);
    decomp_kernel<<<(NE + 255) / 256, 256, 0, stream>>>(X, BUF1f);

    // FFN: relu(x @ W1^T) -> bf16 hidden ; hidden @ W2^T -> fp32 y
    launch_cvt(XBF, X, NE, stream);
    launch_gemm(XBF, W1_l, nullptr, HIDBF, BB * LL, DFF, DDIM, DFF, DFF,
                1, 1, stream);
    launch_gemm(HIDBF, W2_l, nullptr, BUF1f, BB * LL, DDIM, DFF, DDIM, DDIM,
                0, 0, stream);

    // x = (x + y) - mov_avg(x + y)
    add_kernel<<<(NE + 255) / 256, 256, 0, stream>>>(BUF2f, X, BUF1f);
    decomp_kernel<<<(NE + 255) / 256, 256, 0, stream>>>(X, BUF2f);
  }

  pool_kernel<<<(BB * DDIM + 255) / 256, 256, 0, stream>>>(X, POOL);
  head_kernel<<<1, 128, 0, stream>>>(POOL, hw, hb, out);
}
